// LeftTree_45629732553036
// MI455X (gfx1250) — matrix operand order repaired and checked by reading, not compiled, not run on hardware
//
#include <hip/hip_runtime.h>
#include <hip/hip_bf16.h>

// ---------------------------------------------------------------------------
// TreeLSTM scan on MI455X (gfx1250):
//  - persistent 16-block kernel; weights LDS-resident (async-loaded) for all
//    511 steps; c-state carried in registers + bf16 in global for A-fragments
//  - v_wmma_f32_16x16x32_bf16 for all matmuls (f32 accumulate)
//  - A-fragments (c_{t-1} and gathered emb[:,t]) loaded bf16 straight from L2
//    through explicit addrspace(1) pointers (global_load_b128, not flat)
//  - device-scope atomic barrier between steps, ping-pong bf16 c buffers
//  - MLP head as bf16-WMMA GEMMs with fused bias+relu
// ---------------------------------------------------------------------------

typedef __attribute__((ext_vector_type(16))) __bf16 bf16x16;
typedef __attribute__((ext_vector_type(8)))  __bf16 bf16x8;
typedef __attribute__((ext_vector_type(8)))  float  f32x8;

// Global-address-space views so A-fragment loads lower to global_load_b128
// (FLAT would also consume DScnt and contend with the LDS/B-fragment path).
typedef __attribute__((address_space(1))) const __bf16  gc_bf16;
typedef __attribute__((address_space(1))) const bf16x8  gc_bf16x8;

#define VOCAB   32000
#define BATCH   256
#define DMODEL  256
#define SEQ     512
#define BLK_COLS 80     // 5 gates * 16 columns per block
#define NBLOCKS  16

__device__ __forceinline__ float sigmoidf_(float x) { return 1.f / (1.f + __expf(-x)); }

__device__ __forceinline__ f32x8 zero8() {
  f32x8 z = {0.f,0.f,0.f,0.f,0.f,0.f,0.f,0.f};
  return z;
}

// LDS byte offset of a generic pointer into shared memory (AS3 ptrtoint).
__device__ __forceinline__ unsigned lds_offset_of(void* p) {
  return (unsigned)(unsigned long long)(__attribute__((address_space(3))) char*)(char*)p;
}

// CDNA5 async global->LDS copy (16B per lane), tracked by ASYNCcnt.
__device__ __forceinline__ void async_g2l_b128(unsigned lds_byte_off, const void* gaddr) {
  asm volatile("global_load_async_to_lds_b128 %0, %1, off"
               :: "v"(lds_byte_off), "v"(gaddr) : "memory");
}
__device__ __forceinline__ void wait_async0() {
  asm volatile("s_wait_asynccnt 0" ::: "memory");
}

// A fragment (16-bit A 16x32, wave32) loaded from a bf16 row pointer in
// global memory: lane l holds row M = l&15; vector halves are
// K[kb+8h..+7] and K[kb+16+8h..+7] with h = l>>4.
__device__ __forceinline__ bf16x16 load_a_frag_g(gc_bf16* rowp, int kb, int h) {
  gc_bf16* p = rowp + kb + 8 * h;
  bf16x8 lo = *(gc_bf16x8*)(p);
  bf16x8 hi = *(gc_bf16x8*)(p + 16);
  bf16x16 r;
#pragma unroll
  for (int i = 0; i < 8; ++i) { r[i] = lo[i]; r[i + 8] = hi[i]; }
  return r;
}

// Same fragment shape from an LDS-staged [row][k] tile (generic ptr -> ds_load).
__device__ __forceinline__ bf16x16 load_a_frag(const __bf16* base, int rowStride,
                                               int row, int kb, int h) {
  const __bf16* p = base + row * rowStride + kb + 8 * h;
  bf16x8 lo = *(const bf16x8*)(p);
  bf16x8 hi = *(const bf16x8*)(p + 16);
  bf16x16 r;
#pragma unroll
  for (int i = 0; i < 8; ++i) { r[i] = lo[i]; r[i + 8] = hi[i]; }
  return r;
}

// B fragment (16-bit B 32x16, wave32): lane l holds col N = l&15,
// K[kb+8h..+7] and K[kb+16+8h..+7] (h = l>>4), B stored col-major [n][k].
__device__ __forceinline__ bf16x16 load_b_frag(const __bf16* base, int colStride,
                                               int col, int kb, int h) {
  const __bf16* p = base + col * colStride + kb + 8 * h;
  bf16x8 lo = *(const bf16x8*)(p);
  bf16x8 hi = *(const bf16x8*)(p + 16);
  bf16x16 r;
#pragma unroll
  for (int i = 0; i < 8; ++i) { r[i] = lo[i]; r[i + 8] = hi[i]; }
  return r;
}

__device__ __forceinline__ f32x8 wmma_bf16(bf16x16 a, bf16x16 b, f32x8 c) {
  return __builtin_amdgcn_wmma_f32_16x16x32_bf16(false, a, false, b, (short)0, c,
                                                 false, false);
}

// ---------------------------------------------------------------------------
// Repacking kernels (f32 -> bf16)
// ---------------------------------------------------------------------------

// Gate weights [256,1280] row-major -> per-block column-major:
// dst[(bd*80 + g*16 + j)*256 + k] = src[k*1280 + g*256 + bd*16 + j]
__global__ void convert_gate_w(const float* __restrict__ src, __bf16* __restrict__ dst) {
  for (int e = blockIdx.x * blockDim.x + threadIdx.x; e < 256 * 1280;
       e += gridDim.x * blockDim.x) {
    int bd  = e / (BLK_COLS * DMODEL);
    int rem = e % (BLK_COLS * DMODEL);
    int lc  = rem >> 8;
    int k   = rem & 255;
    int g = lc >> 4, j = lc & 15;
    dst[e] = (__bf16)src[k * 1280 + g * 256 + bd * 16 + j];
  }
}

// [K,N] row-major f32 -> [N][K] col-major bf16
__global__ void convert_colmajor(const float* __restrict__ src, __bf16* __restrict__ dst,
                                 int K, int N) {
  int total = K * N;
  for (int e = blockIdx.x * blockDim.x + threadIdx.x; e < total;
       e += gridDim.x * blockDim.x) {
    int n = e / K, k = e % K;
    dst[e] = (__bf16)src[k * N + n];
  }
}

// Straight f32 -> bf16 copy (embedding table)
__global__ void convert_flat(const float* __restrict__ src, __bf16* __restrict__ dst,
                             int total) {
  for (int e = blockIdx.x * blockDim.x + threadIdx.x; e < total;
       e += gridDim.x * blockDim.x)
    dst[e] = (__bf16)src[e];
}

// ---------------------------------------------------------------------------
// Persistent TreeLSTM scan: 16 blocks x 256 threads, one block per WGP.
// Block bd owns gate columns {g*256 + bd*16 + j : g<5, j<16} for all 256 rows.
// LDS: Wl(40KB) + Wr(40KB) only. c-state: registers + bf16 ping-pong global.
// ---------------------------------------------------------------------------
__global__ __launch_bounds__(256, 1) void treelstm_persistent(
    const int* __restrict__ tokens, const float* __restrict__ embed,
    const __bf16* __restrict__ embbf,
    const __bf16* __restrict__ Wlb, const __bf16* __restrict__ Wrb,
    const float* __restrict__ bl,
    __bf16* __restrict__ cbf0, __bf16* __restrict__ cbf1,
    float* __restrict__ hout, unsigned* __restrict__ bar) {
  extern __shared__ char smem_raw[];
  __bf16* sWl = (__bf16*)smem_raw;                 // [80][256]
  __bf16* sWr = sWl + BLK_COLS * DMODEL;           // [80][256]

  const int tid = threadIdx.x;
  const int bd  = blockIdx.x;     // which 16-wide d-tile
  const int w   = tid >> 5;       // wave 0..7
  const int l   = tid & 31;
  const int h   = l >> 4;
  const int ln  = l & 15;

  // --- one-time weight staging via CDNA5 async global->LDS (ASYNCcnt) ---
  {
    const char* g0 = (const char*)(Wlb + (size_t)bd * BLK_COLS * DMODEL);
    const char* g1 = (const char*)(Wrb + (size_t)bd * BLK_COLS * DMODEL);
    unsigned l0 = lds_offset_of(sWl);
    unsigned l1 = lds_offset_of(sWr);
    // 40960 B each = 2560 x 16B
    for (int i = tid; i < 2560; i += 256) {
      async_g2l_b128(l0 + i * 16, g0 + (size_t)i * 16);
      async_g2l_b128(l1 + i * 16, g1 + (size_t)i * 16);
    }
    wait_async0();
    __syncthreads();
  }

  const int dcol = bd * 16 + ln;  // hidden index owned by this lane's column
  float biasr[5];
#pragma unroll
  for (int g = 0; g < 5; ++g) biasr[g] = bl[g * DMODEL + dcol];

  // --- seed: left operand of step 1 is emb[:,0]; each block seeds its own
  // 16 columns of cbf0 and its register-resident c values.
  float creg[2][8];
#pragma unroll
  for (int mt = 0; mt < 2; ++mt) {
#pragma unroll
    for (int r = 0; r < 8; ++r) {
      int row = w * 32 + mt * 16 + r + 8 * h;
      int tk = tokens[row * SEQ];
      float v = embed[(size_t)tk * DMODEL + dcol];
      creg[mt][r] = v;
      cbf0[row * DMODEL + dcol] = (__bf16)v;
    }
  }

  gc_bf16* cbg[2] = {(gc_bf16*)cbf0, (gc_bf16*)cbf1};
  __bf16*  cb[2]  = {cbf0, cbf1};
  gc_bf16* embg   = (gc_bf16*)embbf;
  int epoch = 0;

  // device-scope barrier across the 16 resident workgroups (cumulative count)
  auto grid_barrier = [&]() {
    ++epoch;
    __syncthreads();
    if (tid == 0) {
      __hip_atomic_fetch_add(bar, 1u, __ATOMIC_ACQ_REL, __HIP_MEMORY_SCOPE_AGENT);
      unsigned tgt = (unsigned)epoch * NBLOCKS;
      while (__hip_atomic_load(bar, __ATOMIC_ACQUIRE, __HIP_MEMORY_SCOPE_AGENT) < tgt)
        __builtin_amdgcn_s_sleep(8);
    }
    __syncthreads();
    (void)__hip_atomic_load(bar, __ATOMIC_ACQUIRE, __HIP_MEMORY_SCOPE_AGENT);
  };

  grid_barrier();  // all blocks see seeded cbf0

  for (int t = 1; t < SEQ; ++t) {
    gc_bf16* cprev = cbg[(t - 1) & 1];
    __bf16*  cnext = cb[t & 1];

    f32x8 acc[2][5];
#pragma unroll
    for (int mt = 0; mt < 2; ++mt)
#pragma unroll
      for (int g = 0; g < 5; ++g) acc[mt][g] = zero8();

    // Per-lane A-row pointers (addrspace(1)): c_{t-1} rows and
    // token-gathered emb[:,t] rows.
    gc_bf16* crow0 = cprev + (w * 32 + ln) * DMODEL;
    gc_bf16* crow1 = cprev + (w * 32 + 16 + ln) * DMODEL;
    int tokA0 = tokens[(w * 32 + ln) * SEQ + t];
    int tokA1 = tokens[(w * 32 + 16 + ln) * SEQ + t];
    gc_bf16* xrow0 = embg + (size_t)tokA0 * DMODEL;
    gc_bf16* xrow1 = embg + (size_t)tokA1 * DMODEL;

#pragma unroll
    for (int ks = 0; ks < DMODEL; ks += 32) {
      bf16x16 ac0 = load_a_frag_g(crow0, ks, h);
      bf16x16 ac1 = load_a_frag_g(crow1, ks, h);
      bf16x16 ax0 = load_a_frag_g(xrow0, ks, h);
      bf16x16 ax1 = load_a_frag_g(xrow1, ks, h);
#pragma unroll
      for (int g = 0; g < 5; ++g) {
        bf16x16 blf = load_b_frag(sWl, DMODEL, g * 16 + ln, ks, h);
        bf16x16 brf = load_b_frag(sWr, DMODEL, g * 16 + ln, ks, h);
        acc[0][g] = wmma_bf16(ac0, blf, acc[0][g]);
        acc[0][g] = wmma_bf16(ax0, brf, acc[0][g]);
        acc[1][g] = wmma_bf16(ac1, blf, acc[1][g]);
        acc[1][g] = wmma_bf16(ax1, brf, acc[1][g]);
      }
    }

    // Elementwise cell update. C/D layout: M = r + 8*(l>>4), N = l&15.
    // c_left comes from registers (this lane owns (row,dcol) every step).
#pragma unroll
    for (int mt = 0; mt < 2; ++mt) {
#pragma unroll
      for (int r = 0; r < 8; ++r) {
        int row = w * 32 + mt * 16 + r + 8 * h;
        float gi  = acc[mt][0][r] + biasr[0];
        float gfl = acc[mt][1][r] + biasr[1];
        float gfr = acc[mt][2][r] + biasr[2];
        float gu  = acc[mt][3][r] + biasr[3];
        int tk2 = tokens[row * SEQ + t];
        float xr = embed[(size_t)tk2 * DMODEL + dcol];
        float cn = sigmoidf_(gfl) * creg[mt][r] + sigmoidf_(gfr) * xr +
                   sigmoidf_(gi) * tanhf(gu);
        creg[mt][r] = cn;
        cnext[row * DMODEL + dcol] = (__bf16)cn;
        if (t == SEQ - 1) {
          float go = acc[mt][4][r] + biasr[4];
          hout[row * DMODEL + dcol] = sigmoidf_(go) * tanhf(cn);
        }
      }
    }

    grid_barrier();
  }
}

// ---------------------------------------------------------------------------
// MLP head GEMM: C = relu(A_f32[M,K] @ B_bf16[N][K] + bias), bf16 WMMA.
// Block = 64x32 output tile, 8 waves, each wave one 16x16 tile.
// ---------------------------------------------------------------------------
#define GMT 64
#define GNT 32
#define GKC 64

__global__ __launch_bounds__(256, 2) void gemm_bf16_relu(
    const float* __restrict__ A, const __bf16* __restrict__ B,
    const float* __restrict__ bias, float* __restrict__ C,
    int M, int N, int K, int do_relu) {
  __shared__ __bf16 sA[GMT * GKC];   // [row][k]
  __shared__ __bf16 sB[GNT * GKC];   // [col][k]
  const int tid = threadIdx.x, w = tid >> 5, l = tid & 31, h = l >> 4, ln = l & 15;
  const int mBase = blockIdx.y * GMT, nBase = blockIdx.x * GNT;
  const int mt = w & 3, nt = w >> 2;
  f32x8 acc = zero8();
  for (int kc = 0; kc < K; kc += GKC) {
    __syncthreads();
#pragma unroll
    for (int i = 0; i < 16; ++i) {         // 64*64 f32 -> bf16
      int e = tid + i * 256;
      int row = e >> 6, k = e & 63;
      sA[e] = (__bf16)A[(size_t)(mBase + row) * K + kc + k];
    }
    {                                       // 32*64 bf16 copy (col-major)
      int col = tid >> 3, part = tid & 7;
      const int4* s = (const int4*)(B + (size_t)(nBase + col) * K + kc);
      ((int4*)(sB + col * GKC))[part] = s[part];
    }
    __syncthreads();
#pragma unroll
    for (int ks = 0; ks < GKC; ks += 32) {
      bf16x16 a = load_a_frag(sA, GKC, mt * 16 + ln, ks, h);
      bf16x16 b = load_b_frag(sB, GKC, nt * 16 + ln, ks, h);
      acc = wmma_bf16(a, b, acc);
    }
  }
  const int col = nBase + nt * 16 + ln;
  float bv = bias[col];
#pragma unroll
  for (int r = 0; r < 8; ++r) {
    int row = mBase + mt * 16 + r + 8 * h;
    float v = acc[r] + bv;
    if (do_relu) v = fmaxf(v, 0.f);
    C[(size_t)row * N + col] = v;
  }
}

// Final projection: [256,1024] @ [1024,3] + b2 -> [256,3] (trivial)
__global__ void final_linear(const float* __restrict__ z, const float* __restrict__ W2,
                             const float* __restrict__ b2, float* __restrict__ out) {
  int row = threadIdx.x;
  float a0 = b2[0], a1 = b2[1], a2 = b2[2];
  for (int k = 0; k < 1024; ++k) {
    float v = z[row * 1024 + k];
    a0 += v * W2[k * 3 + 0];
    a1 += v * W2[k * 3 + 1];
    a2 += v * W2[k * 3 + 2];
  }
  out[row * 3 + 0] = a0;
  out[row * 3 + 1] = a1;
  out[row * 3 + 2] = a2;
}

// ---------------------------------------------------------------------------
extern "C" void kernel_launch(void* const* d_in, const int* in_sizes, int n_in,
                              void* d_out, int out_size, void* d_ws, size_t ws_size,
                              hipStream_t stream) {
  (void)in_sizes; (void)n_in; (void)out_size; (void)ws_size;
  const int*   tokens = (const int*)  d_in[0];
  const float* embed  = (const float*)d_in[1];
  const float* W_l    = (const float*)d_in[2];
  const float* b_l    = (const float*)d_in[3];
  const float* W_r    = (const float*)d_in[4];
  const float* W0     = (const float*)d_in[5];
  const float* b0     = (const float*)d_in[6];
  const float* W1     = (const float*)d_in[7];
  const float* b1     = (const float*)d_in[8];
  const float* W2     = (const float*)d_in[9];
  const float* b2     = (const float*)d_in[10];
  float* out = (float*)d_out;

  char* ws = (char*)d_ws;
  size_t off = 0;
  auto carve = [&](size_t bytes) -> char* {
    char* p = ws + off;
    off += (bytes + 255) & ~(size_t)255;
    return p;
  };
  unsigned* bar  = (unsigned*)carve(256);
  __bf16* cbf0   = (__bf16*)carve(256 * 256 * 2);
  __bf16* cbf1   = (__bf16*)carve(256 * 256 * 2);
  float* hbuf    = (float*)carve(256 * 256 * 4);
  float* z0      = (float*)carve(256 * 1024 * 4);
  float* z1      = (float*)carve(256 * 1024 * 4);
  __bf16* Wlb    = (__bf16*)carve(256 * 1280 * 2);
  __bf16* Wrb    = (__bf16*)carve(256 * 1280 * 2);
  __bf16* W0b    = (__bf16*)carve(256 * 1024 * 2);
  __bf16* W1b    = (__bf16*)carve(1024 * 1024 * 2);
  __bf16* embbf  = (__bf16*)carve((size_t)VOCAB * DMODEL * 2);

  hipMemsetAsync(bar, 0, 256, stream);

  convert_gate_w<<<320, 256, 0, stream>>>(W_l, Wlb);
  convert_gate_w<<<320, 256, 0, stream>>>(W_r, Wrb);
  convert_colmajor<<<256, 256, 0, stream>>>(W0, W0b, 256, 1024);
  convert_colmajor<<<1024, 256, 0, stream>>>(W1, W1b, 1024, 1024);
  convert_flat<<<2048, 256, 0, stream>>>(embed, embbf, VOCAB * DMODEL);

  const size_t lds_bytes = (size_t)(2 * BLK_COLS * DMODEL) * sizeof(__bf16);  // 80 KB
  hipFuncSetAttribute((const void*)treelstm_persistent,
                      hipFuncAttributeMaxDynamicSharedMemorySize, (int)lds_bytes);
  treelstm_persistent<<<NBLOCKS, 256, lds_bytes, stream>>>(
      tokens, embed, embbf, Wlb, Wrb, b_l, cbf0, cbf1, hbuf, bar);

  gemm_bf16_relu<<<dim3(1024 / GNT, 256 / GMT), 256, 0, stream>>>(
      hbuf, W0b, b0, z0, 256, 1024, 256, 1);
  gemm_bf16_relu<<<dim3(1024 / GNT, 256 / GMT), 256, 0, stream>>>(
      z0, W1b, b1, z1, 256, 1024, 1024, 1);
  final_linear<<<1, 256, 0, stream>>>(z1, W2, b2, out);
}
